// Block_2388001817000
// MI455X (gfx1250) — matrix operand order repaired and checked by reading, not compiled, not run on hardware
//
#include <hip/hip_runtime.h>
#include <math.h>

// ---------------------------------------------------------------------------
// CDNA5 (gfx1250) transformer block: pre-LN attention (softmax1) + pre-LN MLP.
// GEMMs: TDM (tensor_load_to_lds) double-buffered LDS staging + WMMA f16.
// Attention: fused flash-style softmax1, bit-packed mask, 32 q-rows/wave.
// Shapes fixed by the reference: B=4, N=1024, C=1024, H=16, D=64, Hd=4096.
// ---------------------------------------------------------------------------

typedef _Float16 half_t;
typedef __attribute__((ext_vector_type(16))) _Float16 v16h;
typedef __attribute__((ext_vector_type(8)))  _Float16 v8h;
typedef __attribute__((ext_vector_type(8)))  float    v8f;
typedef __attribute__((ext_vector_type(4)))  unsigned int u32x4;
typedef __attribute__((ext_vector_type(8)))  int i32x8;
typedef __attribute__((ext_vector_type(4)))  int i32x4;

__device__ __forceinline__ v16h cat8(v8h lo, v8h hi) {
  return __builtin_shufflevector(lo, hi, 0,1,2,3,4,5,6,7,8,9,10,11,12,13,14,15);
}
__device__ __forceinline__ v8f wmma_f16(v16h a, v16h b, v8f c) {
  return __builtin_amdgcn_wmma_f32_16x16x32_f16(false, a, false, b, (short)0, c,
                                                false, false);
}
__device__ __forceinline__ unsigned lds_off(const void* p) {
  // generic pointers to LDS carry the LDS byte offset in the low 32 bits
  return (unsigned)(size_t)p;
}

// ---------------------------------------------------------------------------
// TDM: 2D tile load Global -> LDS.  tile = rows x 32 halves, data_size = 2B,
// LDS padding: +4 dwords every 16 dwords => 80B row stride (40 halves).
// D# packing per CDNA5 ISA ss8.3/8.4. This toolchain's
// builtin takes 6 args: (g0, g1, g2, g3, extra, cpol); groups 2/3 and the
// trailing group are unused for a 2D non-cluster descriptor -> zeros.
// ---------------------------------------------------------------------------
__device__ __forceinline__ void tdm_load_tile32(const half_t* g, unsigned ldsaddr,
                                                unsigned Kelems, unsigned rows) {
  unsigned long long ga = (unsigned long long)(size_t)g;
  u32x4 g0;
  g0[0] = 1u;                                     // count=1, user-mode, no gather
  g0[1] = ldsaddr;                                // lds_addr (bytes)
  g0[2] = (unsigned)(ga & 0xffffffffu);           // global_addr[31:0]
  g0[3] = (unsigned)((ga >> 32) & 0x1ffffffu)     // global_addr[56:32]
          | (2u << 30);                           // type = 2 ("image")
  i32x8 g1;
  unsigned w0 = (1u << 16)                        // data_size = 2 bytes
              | (1u << 20)                        // pad_enable
              | (3u << 22)                        // pad_interval: 16 dwords
              | (3u << 25);                       // pad_amount: 4 dwords
  g1[0] = (int)w0;
  g1[1] = (int)((Kelems & 0xffffu) << 16);        // tensor_dim0[15:0] @ [63:48]
  g1[2] = (int)((Kelems >> 16) | (rows << 16));   // dim0[31:16] | tensor_dim1[15:0]
  g1[3] = (int)((32u << 16));                     // dim1[31:16]=0 | tile_dim0=32
  g1[4] = (int)(rows & 0xffffu);                  // tile_dim1 | tile_dim2=0
  g1[5] = (int)Kelems;                            // tensor_dim0_stride[31:0]
  g1[6] = 0;                                      // stride0[47:32] | stride1[15:0]
  g1[7] = 0;                                      // stride1[47:16]
  i32x4 z4 = {0, 0, 0, 0};                        // groups 2/3 unused (2D tile)
  i32x8 z8 = {0, 0, 0, 0, 0, 0, 0, 0};            // trailing group unused
  __builtin_amdgcn_tensor_load_to_lds(g0, g1, z4, z4, z8, 0);
}

// ---------------------------------------------------------------------------
// fp32 -> fp16 conversion (weights)
// ---------------------------------------------------------------------------
__global__ void cvt_f32_to_f16(const float* __restrict__ in,
                               half_t* __restrict__ out, int n) {
  int i = blockIdx.x * blockDim.x + threadIdx.x;
  if (i < n) out[i] = (half_t)in[i];
}

// ---------------------------------------------------------------------------
// Pack bool mask [B,N,N] into bits [B,N,N/32] (bit j = key j masked)
// ---------------------------------------------------------------------------
__global__ void pack_mask_bits(const unsigned char* __restrict__ mask,
                               unsigned* __restrict__ bits, int total) {
  int i = blockIdx.x * blockDim.x + threadIdx.x;
  if (i >= total) return;
  const unsigned char* p = mask + (size_t)i * 32;
  unsigned w = 0;
#pragma unroll
  for (int j = 0; j < 32; ++j) w |= (p[j] ? 1u : 0u) << j;
  bits[i] = w;
}

// ---------------------------------------------------------------------------
// Row LayerNorm (C=1024) -> fp16 output. One 256-thread block per row.
// ---------------------------------------------------------------------------
__global__ __launch_bounds__(256) void layernorm_f16(
    const float* __restrict__ x, const float* __restrict__ gamma,
    const float* __restrict__ beta, half_t* __restrict__ out) {
  const int C = 1024;
  int row = blockIdx.x, tid = threadIdx.x;
  const float* xr = x + (size_t)row * C;
  float v[4], s = 0.f, s2 = 0.f;
#pragma unroll
  for (int i = 0; i < 4; ++i) {
    v[i] = xr[tid + 256 * i];
    s += v[i];
    s2 += v[i] * v[i];
  }
#pragma unroll
  for (int m = 16; m >= 1; m >>= 1) {
    s  += __shfl_xor(s, m);
    s2 += __shfl_xor(s2, m);
  }
  __shared__ float ws[8], ws2[8], stat[2];
  if ((tid & 31) == 0) { ws[tid >> 5] = s; ws2[tid >> 5] = s2; }
  __syncthreads();
  if (tid == 0) {
    float a = 0.f, a2 = 0.f;
#pragma unroll
    for (int j = 0; j < 8; ++j) { a += ws[j]; a2 += ws2[j]; }
    float mu  = a / (float)C;
    float var = a2 / (float)C - mu * mu;
    stat[0] = mu;
    stat[1] = rsqrtf(var + 1e-5f);
  }
  __syncthreads();
  float mu = stat[0], rs = stat[1];
  half_t* orow = out + (size_t)row * C;
#pragma unroll
  for (int i = 0; i < 4; ++i) {
    int c = tid + 256 * i;
    orow[c] = (half_t)((v[i] - mu) * rs * gamma[c] + beta[c]);
  }
}

// ---------------------------------------------------------------------------
// WMMA GEMM with TDM double-buffered LDS staging.
// Block = 256 threads (8 waves, 2x4), block tile 128(m) x 256(n), K step 32.
// Wave 0 issues tensor_load_to_lds for the next K-slice; TENSORcnt is
// in-order per wave so s_wait_tensorcnt(2) guarantees the current slice is
// resident before the barrier releases the consumers.
// out[M,N] = A[M,K]f16 @ B[N,K]^T f16 + bias (+GELU) (+res); f32/f16 outputs.
// ---------------------------------------------------------------------------
template <int ACT>  // 0 = none, 1 = exact GELU
__global__ __launch_bounds__(256) void gemm_wmma_f16(
    const half_t* __restrict__ A, const half_t* __restrict__ Bm,
    const float* __restrict__ bias, const float* __restrict__ res,
    float* __restrict__ out32, half_t* __restrict__ out16,
    int M, int Nn, int K) {
  const int LDT = 40;                       // padded halves per 32-half row
  __shared__ __align__(16) half_t lds[2][384 * LDT];  // A:128 rows + B:256 rows
  int lane = threadIdx.x & 31, wv = threadIdx.x >> 5;
  int tiles_n = Nn >> 8;
  int tm = blockIdx.x / tiles_n, tn = blockIdx.x % tiles_n;
  int wm = wv >> 2, wn = wv & 3;
  int lo = lane & 15, hf = lane >> 4;

  const half_t* Ag = A  + (size_t)tm * 128 * K;
  const half_t* Bg = Bm + (size_t)tn * 256 * K;

  if (wv == 0) {  // TDM ignores EXEC: guard at wave granularity
    tdm_load_tile32(Ag, lds_off(&lds[0][0]),         (unsigned)K, 128u);
    tdm_load_tile32(Bg, lds_off(&lds[0][128 * LDT]), (unsigned)K, 256u);
  }

  v8f acc[4][4] = {};
  int nsteps = K >> 5;
  for (int st = 0; st < nsteps; ++st) {
    int p = st & 1;
    if (wv == 0) {
      if (st + 1 < nsteps) {
        tdm_load_tile32(Ag + (size_t)(st + 1) * 32,
                        lds_off(&lds[p ^ 1][0]), (unsigned)K, 128u);
        tdm_load_tile32(Bg + (size_t)(st + 1) * 32,
                        lds_off(&lds[p ^ 1][128 * LDT]), (unsigned)K, 256u);
        __builtin_amdgcn_s_wait_tensorcnt((short)2);
      } else {
        __builtin_amdgcn_s_wait_tensorcnt((short)0);
      }
    }
    __syncthreads();
    const half_t* Ab = &lds[p][0];
    const half_t* Bb = &lds[p][128 * LDT];
    v16h af[4], bf[4];
#pragma unroll
    for (int i = 0; i < 4; ++i) {
      const half_t* pa = Ab + (size_t)(wm * 64 + i * 16 + lo) * LDT + hf * 8;
      af[i] = cat8(*(const v8h*)pa, *(const v8h*)(pa + 16));
    }
#pragma unroll
    for (int j = 0; j < 4; ++j) {
      const half_t* pb = Bb + (size_t)(wn * 64 + j * 16 + lo) * LDT + hf * 8;
      bf[j] = cat8(*(const v8h*)pb, *(const v8h*)(pb + 16));
    }
#pragma unroll
    for (int i = 0; i < 4; ++i)
#pragma unroll
      for (int j = 0; j < 4; ++j)
        acc[i][j] = wmma_f16(af[i], bf[j], acc[i][j]);
    __syncthreads();
  }

  // Epilogue. C-frag: VGPR r, lanes 0-15 -> M=r, lanes 16-31 -> M=8+r.
#pragma unroll
  for (int i = 0; i < 4; ++i) {
#pragma unroll
    for (int r = 0; r < 8; ++r) {
      int m = tm * 128 + wm * 64 + i * 16 + r + hf * 8;
#pragma unroll
      for (int j = 0; j < 4; ++j) {
        int n = tn * 256 + wn * 64 + j * 16 + lo;
        float v = acc[i][j][r];
        if (bias) v += bias[n];
        if (ACT == 1) v = 0.5f * v * (1.0f + erff(v * 0.70710678118654752f));
        if (res) v += res[(size_t)m * Nn + n];
        if (out32) out32[(size_t)m * Nn + n] = v;
        if (out16) out16[(size_t)m * Nn + n] = (half_t)v;
      }
    }
  }
}

// ---------------------------------------------------------------------------
// Split fused qkv[B,N,3,H,D] (f16) into q[B,H,N,D], k[B,H,N,D], vT[B,H,D,N]
// ---------------------------------------------------------------------------
__global__ void split_qkv(const half_t* __restrict__ qkv,
                          half_t* __restrict__ q, half_t* __restrict__ k,
                          half_t* __restrict__ vT) {
  size_t i = (size_t)blockIdx.x * blockDim.x + threadIdx.x;  // B*N*H*D threads
  int d = (int)(i & 63);
  size_t t = i >> 6;
  int h = (int)(t & 15); t >>= 4;
  int n = (int)(t & 1023);
  int b = (int)(t >> 10);
  const half_t* src = qkv + ((size_t)(b * 1024 + n)) * 3072 + h * 64 + d;
  size_t qi = (((size_t)(b * 16 + h) * 1024 + n) * 64) + d;
  q[qi] = src[0];
  k[qi] = src[1024];
  vT[(((size_t)(b * 16 + h) * 64 + d) * 1024) + n] = src[2048];
}

// ---------------------------------------------------------------------------
// Fused attention with softmax1 + bit-packed mask. One wave per 32 query
// rows of one (b,h). Online softmax init m=0,l=1 == implicit zero logit.
// P relayout C-frag -> A-frag via LDS (DS ops in-order per wave; explicit
// s_wait_dscnt 0 covers the cross-lane store->load).
// ---------------------------------------------------------------------------
__global__ __launch_bounds__(128) void attn_softmax1(
    const half_t* __restrict__ q16, const half_t* __restrict__ k16,
    const half_t* __restrict__ vT16, const unsigned* __restrict__ maskbits,
    half_t* __restrict__ o16) {
  const int N = 1024, Dh = 64, H = 16;
  int lane = threadIdx.x & 31, wv = threadIdx.x >> 5;
  int tile = blockIdx.x * 4 + wv;  // 0..2047
  int qt = tile & 31;              // 32-row query block
  int h  = (tile >> 5) & 15;
  int b  = tile >> 9;
  int lo = lane & 15, hf = lane >> 4;

  const half_t* Q  = q16  + ((size_t)(b * H + h)) * N * Dh;
  const half_t* Kp = k16  + ((size_t)(b * H + h)) * N * Dh;
  const half_t* Vt = vT16 + ((size_t)(b * H + h)) * Dh * N;
  const unsigned* Mb = maskbits + (size_t)b * N * (N / 32);

  __shared__ __align__(16) half_t smem[4][2 * 16 * 40];
  half_t* sp = smem[wv];

  // Q A-fragments: 2 M-tiles x 2 K(=d) chunks
  v16h qa[2][2];
#pragma unroll
  for (int mi = 0; mi < 2; ++mi) {
    int qrow = qt * 32 + mi * 16 + lo;
#pragma unroll
    for (int kk = 0; kk < 2; ++kk) {
      const half_t* p = Q + (size_t)qrow * Dh + kk * 32 + hf * 8;
      qa[mi][kk] = cat8(*(const v8h*)p, *(const v8h*)(p + 16));
    }
  }

  v8f oa[2][4] = {};
  float mrow[2][8], lrow[2][8];
#pragma unroll
  for (int mi = 0; mi < 2; ++mi)
#pragma unroll
    for (int r = 0; r < 8; ++r) { mrow[mi][r] = 0.f; lrow[mi][r] = 1.f; }

  const float scale = 0.125f;            // D^-0.5, D=64
  int maskrow = qt * 32 + lane;          // this lane serves row `lane`

  for (int kt = 0; kt < N; kt += 32) {
    unsigned wmask = Mb[(size_t)maskrow * (N / 32) + (kt >> 5)];
    // S tiles: [mi][t] covers queries mi*16.. and keys kt+t*16..
    v8f s[2][2];
#pragma unroll
    for (int t = 0; t < 2; ++t) {
      int key = kt + t * 16 + lo;
      const half_t* kp0 = Kp + (size_t)key * Dh + hf * 8;
      v16h bf0 = cat8(*(const v8h*)kp0, *(const v8h*)(kp0 + 16));
      v16h bf1 = cat8(*(const v8h*)(kp0 + 32), *(const v8h*)(kp0 + 48));
#pragma unroll
      for (int mi = 0; mi < 2; ++mi) {
        v8f c = {};
        c = wmma_f16(qa[mi][0], bf0, c);
        c = wmma_f16(qa[mi][1], bf1, c);
        s[mi][t] = c;
      }
    }
    // scale + mask + online softmax; row r of tile mi lives in one lane-half
#pragma unroll
    for (int mi = 0; mi < 2; ++mi) {
#pragma unroll
      for (int r = 0; r < 8; ++r) {
        int rl = mi * 16 + r + hf * 8;                 // local row 0..31
        unsigned wr = (unsigned)__shfl((int)wmask, rl);
        float s0 = s[mi][0][r] * scale;
        float s1 = s[mi][1][r] * scale;
        if ((wr >> lo) & 1u)        s0 = -3.0e38f;
        if ((wr >> (lo + 16)) & 1u) s1 = -3.0e38f;
        float mx = fmaxf(s0, s1);
#pragma unroll
        for (int dm = 8; dm >= 1; dm >>= 1) mx = fmaxf(mx, __shfl_xor(mx, dm));
        float mn   = fmaxf(mrow[mi][r], mx);
        float corr = __expf(mrow[mi][r] - mn);
        float p0 = __expf(s0 - mn), p1 = __expf(s1 - mn);
        float ps = p0 + p1;
#pragma unroll
        for (int dm = 8; dm >= 1; dm >>= 1) ps += __shfl_xor(ps, dm);
        lrow[mi][r] = lrow[mi][r] * corr + ps;
        mrow[mi][r] = mn;
#pragma unroll
        for (int f = 0; f < 4; ++f) oa[mi][f][r] *= corr;
        int row = r + hf * 8;
        sp[mi * 640 + row * 40 + lo]      = (half_t)p0;
        sp[mi * 640 + row * 40 + 16 + lo] = (half_t)p1;
      }
    }
    asm volatile("s_wait_dscnt 0" ::: "memory");
    v16h pa[2];
#pragma unroll
    for (int mi = 0; mi < 2; ++mi) {
      const half_t* pp = sp + mi * 640 + lo * 40 + hf * 8;
      pa[mi] = cat8(*(const v8h*)pp, *(const v8h*)(pp + 16));
    }
    // O += P @ V (B operand from pre-transposed V^T, contiguous per lane)
#pragma unroll
    for (int f = 0; f < 4; ++f) {
      const half_t* vp = Vt + (size_t)(f * 16 + lo) * N + kt + hf * 8;
      v16h bf = cat8(*(const v8h*)vp, *(const v8h*)(vp + 16));
#pragma unroll
      for (int mi = 0; mi < 2; ++mi) oa[mi][f] = wmma_f16(pa[mi], bf, oa[mi][f]);
    }
  }
  // normalize and write o[b, q, h*64 + d] as f16
#pragma unroll
  for (int mi = 0; mi < 2; ++mi) {
#pragma unroll
    for (int r = 0; r < 8; ++r) {
      float inv = 1.0f / lrow[mi][r];
      int qr = qt * 32 + mi * 16 + r + hf * 8;
      half_t* op = o16 + ((size_t)(b * N + qr)) * 1024 + h * Dh;
#pragma unroll
      for (int f = 0; f < 4; ++f) op[f * 16 + lo] = (half_t)(oa[mi][f][r] * inv);
    }
  }
}

// ---------------------------------------------------------------------------
// Host-side orchestration
// ---------------------------------------------------------------------------
extern "C" void kernel_launch(void* const* d_in, const int* in_sizes, int n_in,
                              void* d_out, int out_size, void* d_ws,
                              size_t ws_size, hipStream_t stream) {
  (void)in_sizes; (void)n_in; (void)out_size; (void)ws_size;
  const int Cc = 1024, HD = 4096, Mr = 4096;  // B*N = 4096 rows

  const float* x      = (const float*)d_in[0];
  const unsigned char* mask = (const unsigned char*)d_in[1];  // jax bool = 1B
  const float* ln1_g  = (const float*)d_in[2];
  const float* ln1_b  = (const float*)d_in[3];
  const float* qkv_w  = (const float*)d_in[4];
  const float* qkv_b  = (const float*)d_in[5];
  const float* proj_w = (const float*)d_in[6];
  const float* proj_b = (const float*)d_in[7];
  const float* ln2_g  = (const float*)d_in[8];
  const float* ln2_b  = (const float*)d_in[9];
  const float* fc1_w  = (const float*)d_in[10];
  const float* fc1_b  = (const float*)d_in[11];
  const float* fc2_w  = (const float*)d_in[12];
  const float* fc2_b  = (const float*)d_in[13];
  float* out = (float*)d_out;

  char* ws = (char*)d_ws;
  size_t off = 0;
  auto alloc = [&](size_t bytes) -> void* {
    void* p = ws + off;
    off += (bytes + 255) & ~(size_t)255;
    return p;
  };
  half_t* h16     = (half_t*)alloc((size_t)Mr * Cc * 2);
  half_t* qkvw16  = (half_t*)alloc((size_t)3 * Cc * Cc * 2);
  half_t* projw16 = (half_t*)alloc((size_t)Cc * Cc * 2);
  half_t* fc1w16  = (half_t*)alloc((size_t)HD * Cc * 2);
  half_t* fc2w16  = (half_t*)alloc((size_t)Cc * HD * 2);
  half_t* qkv16   = (half_t*)alloc((size_t)Mr * 3 * Cc * 2);
  half_t* q16     = (half_t*)alloc((size_t)Mr * Cc * 2);
  half_t* k16     = (half_t*)alloc((size_t)Mr * Cc * 2);
  half_t* vT16    = (half_t*)alloc((size_t)Mr * Cc * 2);
  half_t* o16     = (half_t*)alloc((size_t)Mr * Cc * 2);
  float*  x1      = (float*)alloc((size_t)Mr * Cc * 4);
  half_t* h2_16   = (half_t*)alloc((size_t)Mr * Cc * 2);
  half_t* m16     = (half_t*)alloc((size_t)Mr * HD * 2);
  unsigned* mbits = (unsigned*)alloc((size_t)4 * 1024 * 32 * 4);  // [B,N,N/32]

  // Preprocessing (independent of each other)
  cvt_f32_to_f16<<<(3 * Cc * Cc) / 256, 256, 0, stream>>>(qkv_w, qkvw16, 3 * Cc * Cc);
  cvt_f32_to_f16<<<(Cc * Cc) / 256, 256, 0, stream>>>(proj_w, projw16, Cc * Cc);
  cvt_f32_to_f16<<<(HD * Cc) / 256, 256, 0, stream>>>(fc1_w, fc1w16, HD * Cc);
  cvt_f32_to_f16<<<(Cc * HD) / 256, 256, 0, stream>>>(fc2_w, fc2w16, Cc * HD);
  pack_mask_bits<<<(4 * 1024 * 32) / 256, 256, 0, stream>>>(mask, mbits,
                                                            4 * 1024 * 32);

  // h = LN1(x) -> f16
  layernorm_f16<<<Mr, 256, 0, stream>>>(x, ln1_g, ln1_b, h16);

  // qkv = h @ qkv_w^T + qkv_b  [4096 x 3072] -> f16
  gemm_wmma_f16<0><<<32 * 12, 256, 0, stream>>>(
      h16, qkvw16, qkv_b, nullptr, nullptr, qkv16, Mr, 3 * Cc, Cc);

  // split into q/k (row-major) and v^T
  split_qkv<<<(Mr * Cc) / 256, 256, 0, stream>>>(qkv16, q16, k16, vT16);

  // fused masked softmax1 attention -> o16 [B,N,C]
  attn_softmax1<<<2048 / 4, 128, 0, stream>>>(q16, k16, vT16, mbits, o16);

  // x1 = x + o @ proj_w^T + proj_b
  gemm_wmma_f16<0><<<32 * 4, 256, 0, stream>>>(
      o16, projw16, proj_b, x, x1, nullptr, Mr, Cc, Cc);

  // h2 = LN2(x1) -> f16
  layernorm_f16<<<Mr, 256, 0, stream>>>(x1, ln2_g, ln2_b, h2_16);

  // m = GELU(h2 @ fc1_w^T + fc1_b)  [4096 x 4096] -> f16
  gemm_wmma_f16<1><<<32 * 16, 256, 0, stream>>>(
      h2_16, fc1w16, fc1_b, nullptr, nullptr, m16, Mr, HD, Cc);

  // out = x1 + m @ fc2_w^T + fc2_b
  gemm_wmma_f16<0><<<32 * 4, 256, 0, stream>>>(
      m16, fc2w16, fc2_b, x1, out, nullptr, Mr, Cc, HD);
}
